// MoE_82927228551526
// MI455X (gfx1250) — matrix operand order repaired and checked by reading, not compiled, not run on hardware
//
#include <hip/hip_runtime.h>
#include <cmath>

typedef _Float16 f16;
typedef __attribute__((ext_vector_type(4)))  _Float16 v4h;
typedef __attribute__((ext_vector_type(8)))  _Float16 v8h;
typedef __attribute__((ext_vector_type(16))) _Float16 v16h;
typedef __attribute__((ext_vector_type(8)))  float    v8f;

#define TOKENS   8192
#define DM       1024
#define DF       4096
#define NE       8
#define TAU      2.2f

#define MT   64                  // tokens per block
#define FC   256                 // d_ff chunk width (16 cols per wave in phase 1)
#define XLD2 (DM + 8)            // padded f16 LDS stride, x tile
#define HLD2 (FC + 8)            // padded f16 LDS stride, h chunk

static __device__ __forceinline__ v8f wmma16(v16h a, v16h b, v8f c) {
  // D(16x16 f32) = A(16x32 f16) * B(32x16 f16) + C
  return __builtin_amdgcn_wmma_f32_16x16x32_f16(false, a, false, b, (short)0, c,
                                                false, false);
}

// build a v16h operand from two contiguous 16-byte chunks (b128 loads)
static __device__ __forceinline__ v16h ld16(const f16* p0, const f16* p1) {
  v8h lo = *(const v8h*)p0;
  v8h hi = *(const v8h*)p1;
  return __builtin_shufflevector(lo, hi, 0, 1, 2, 3, 4, 5, 6, 7, 8, 9, 10, 11,
                                 12, 13, 14, 15);
}

// ---- one-time: W[e][K][N] f32 -> Wt[e][N][K] f16 (transpose + convert) ----
__global__ void __launch_bounds__(256)
transpose_cvt_kernel(const float* __restrict__ W, f16* __restrict__ Wt,
                     int K, int N) {
  __shared__ float t[32][33];
  const int e  = blockIdx.z;
  const int n0 = blockIdx.x * 32;
  const int k0 = blockIdx.y * 32;
  const int tx = threadIdx.x & 31;
  const int ty = threadIdx.x >> 5;  // 0..7
  const float* src = W + ((size_t)e * K + k0) * N + n0;
#pragma unroll
  for (int j = 0; j < 4; ++j)
    t[ty + 8 * j][tx] = src[(size_t)(ty + 8 * j) * N + tx];
  __syncthreads();
  f16* dst = Wt + ((size_t)e * N + n0) * K + k0;
#pragma unroll
  for (int j = 0; j < 4; ++j)
    dst[(size_t)(ty + 8 * j) * K + tx] = (f16)t[tx][ty + 8 * j];
}

// ---------------- gating: softmax((x@Wg + bg)/TAU), f32 ----------------
__global__ void __launch_bounds__(256)
gate_kernel(const float* __restrict__ x, const float* __restrict__ Wg,
            const float* __restrict__ bg, float* __restrict__ gates) {
  const int wave = threadIdx.x >> 5;
  const int lane = threadIdx.x & 31;
  const int t = blockIdx.x * 8 + wave;

  float acc[NE] = {};
  const float* xp = x + (size_t)t * DM;
  for (int d = lane; d < DM; d += 32) {
    const float xv = xp[d];
    const float* wg = Wg + (size_t)d * NE;
#pragma unroll
    for (int e = 0; e < NE; ++e) acc[e] += xv * wg[e];
  }
#pragma unroll
  for (int off = 16; off > 0; off >>= 1) {
#pragma unroll
    for (int e = 0; e < NE; ++e) acc[e] += __shfl_xor(acc[e], off, 32);
  }
  float l[NE];
  float mx = -3.4e38f;
#pragma unroll
  for (int e = 0; e < NE; ++e) {
    l[e] = (acc[e] + bg[e]) * (1.0f / TAU);
    mx = fmaxf(mx, l[e]);
  }
  float s = 0.0f;
#pragma unroll
  for (int e = 0; e < NE; ++e) { l[e] = __expf(l[e] - mx); s += l[e]; }
  const float inv = 1.0f / s;
  if (lane == 0) {
#pragma unroll
    for (int e = 0; e < NE; ++e) gates[(size_t)t * NE + e] = l[e] * inv;
  }
}

// ---- fused FFN: out = sum_e g_e * (gelu(x@W1+b1) @ W2 + b2), f16 WMMA ----
__global__ void __launch_bounds__(512)
moe_ffn_kernel(const float* __restrict__ x,
               const f16* __restrict__ W1t,   // [e][f][k]  k contiguous
               const float* __restrict__ b1,
               const f16* __restrict__ W2t,   // [e][d][f]  f contiguous
               const float* __restrict__ b2,
               const float* __restrict__ gates, float* __restrict__ out) {
  extern __shared__ char smem[];
  f16*   x_lds = (f16*)smem;                        // MT * XLD2 f16
  f16*   h_lds0 = x_lds + (size_t)MT * XLD2;        // MT * HLD2 f16 (buf 0)
  f16*   h_lds1 = h_lds0 + (size_t)MT * HLD2;       // MT * HLD2 f16 (buf 1)
  float* g_lds = (float*)(h_lds1 + (size_t)MT * HLD2);  // MT * NE f32

  const int tid  = threadIdx.x;
  const int wave = tid >> 5;        // 0..15
  const int lane = tid & 31;
  const int l15  = lane & 15;
  const int kh   = lane >> 4;       // which K half this lane holds
  const int rb   = kh * 8;          // C-tile row offset for this lane half
  const int tok0 = blockIdx.x * MT;

  // stage x tile [64 x 1024] f32 -> f16 LDS (reused 8 experts * whole d_ff)
  for (int i = tid; i < MT * (DM / 4); i += 512) {
    const int row = i >> 8;  // DM/4 == 256
    const int c4  = i & 255;
    const float4 v =
        reinterpret_cast<const float4*>(x + (size_t)(tok0 + row) * DM)[c4];
    v4h hv = {(f16)v.x, (f16)v.y, (f16)v.z, (f16)v.w};
    *(v4h*)&x_lds[(size_t)row * XLD2 + c4 * 4] = hv;
  }
  for (int i = tid; i < MT * NE; i += 512)
    g_lds[i] = gates[(size_t)tok0 * NE + i];
  __syncthreads();

  // hoisted per-m A pointers into the x tile (rows m*16 + l15)
  const f16* ax[4];
#pragma unroll
  for (int m = 0; m < 4; ++m)
    ax[m] = &x_lds[(size_t)(m * 16 + l15) * XLD2 + kh * 8];

  // out accumulator: wave owns [64 rows x 64 cols] = 4x4 tiles = 128 VGPRs
  v8f acc[4][4];
#pragma unroll
  for (int m = 0; m < 4; ++m)
#pragma unroll
    for (int n = 0; n < 4; ++n) acc[m][n] = {};

  int pb = 0;  // h double-buffer parity: one barrier per chunk
  for (int e = 0; e < NE; ++e) {
    const f16* __restrict__ W1e = W1t + (size_t)e * DF * DM;
    const f16* __restrict__ W2e = W2t + (size_t)e * DM * DF;

    for (int fc = 0; fc < DF; fc += FC) {
      f16* hw = pb ? h_lds1 : h_lds0;
      pb ^= 1;

      // ---- phase 1: wave computes h cols [fc+wave*16, +16) for all 64 rows
      const int col0 = fc + wave * 16;
      v8f h[4];
#pragma unroll
      for (int m = 0; m < 4; ++m) h[m] = {};
      const f16* __restrict__ bptr =
          W1e + ((size_t)(col0 + l15)) * DM + kh * 8;
#pragma unroll 2
      for (int k = 0; k < DM; k += 32) {
        const v16h b = ld16(bptr + k, bptr + k + 16);
        v16h a[4];
#pragma unroll
        for (int m = 0; m < 4; ++m) a[m] = ld16(ax[m] + k, ax[m] + k + 16);
#pragma unroll
        for (int m = 0; m < 4; ++m) h[m] = wmma16(a[m], b, h[m]);
      }
      // bias + exact GELU + fold gate, store h chunk as f16
      const float bias = b1[(size_t)e * DF + col0 + l15];
#pragma unroll
      for (int m = 0; m < 4; ++m) {
#pragma unroll
        for (int i = 0; i < 8; ++i) {
          const int row = m * 16 + rb + i;
          float v = h[m][i] + bias;
          v = 0.5f * v * (1.0f + erff(v * 0.70710678118654752f));
          hw[(size_t)row * HLD2 + wave * 16 + l15] =
              (f16)(v * g_lds[row * NE + e]);
        }
      }
      __syncthreads();  // h chunk fully written (also fences prior buf reads)

      // ---- phase 2: acc += h_chunk[64xFC] @ W2[fc:fc+FC, wave*64:+64]
      const f16* ah[4];
#pragma unroll
      for (int m = 0; m < 4; ++m)
        ah[m] = &hw[(size_t)(m * 16 + l15) * HLD2 + kh * 8];
      const f16* __restrict__ w2p =
          W2e + ((size_t)(wave * 64 + l15)) * DF + fc + kh * 8;
#pragma unroll 2
      for (int k = 0; k < FC; k += 32) {
        v16h a[4];
#pragma unroll
        for (int m = 0; m < 4; ++m) a[m] = ld16(ah[m] + k, ah[m] + k + 16);
#pragma unroll
        for (int n = 0; n < 4; ++n) {
          const f16* bp = w2p + (size_t)n * 16 * DF + k;
          const v16h b = ld16(bp, bp + 16);
#pragma unroll
          for (int m = 0; m < 4; ++m) acc[m][n] = wmma16(a[m], b, acc[m][n]);
        }
      }
      // no trailing barrier: next chunk writes the other h buffer
    }
  }

  // ---- epilogue: out = acc + sum_e g[row,e]*b2[e,col], single store ----
#pragma unroll
  for (int n = 0; n < 4; ++n) {
    const int col = wave * 64 + n * 16 + l15;
    float b2v[NE];
#pragma unroll
    for (int e = 0; e < NE; ++e) b2v[e] = b2[(size_t)e * DM + col];
#pragma unroll
    for (int m = 0; m < 4; ++m) {
#pragma unroll
      for (int i = 0; i < 8; ++i) {
        const int row = m * 16 + rb + i;
        float s = acc[m][n][i];
#pragma unroll
        for (int e = 0; e < NE; ++e) s += g_lds[row * NE + e] * b2v[e];
        out[(size_t)(tok0 + row) * DM + col] = s;
      }
    }
  }
}

extern "C" void kernel_launch(void* const* d_in, const int* in_sizes, int n_in,
                              void* d_out, int out_size, void* d_ws,
                              size_t ws_size, hipStream_t stream) {
  const float* x  = (const float*)d_in[0];
  const float* W1 = (const float*)d_in[1];
  const float* b1 = (const float*)d_in[2];
  const float* W2 = (const float*)d_in[3];
  const float* b2 = (const float*)d_in[4];
  const float* Wg = (const float*)d_in[5];
  const float* bg = (const float*)d_in[6];
  float* out = (float*)d_out;

  // workspace layout: W1t f16 (64 MiB) | W2t f16 (64 MiB) | gates f32 (256 KiB)
  const size_t WELTS = (size_t)NE * DM * DF;  // 33.55M elements each
  f16* W1t = (f16*)d_ws;
  f16* W2t = W1t + WELTS;
  float* gates = (float*)(W2t + WELTS);

  // W1 [e][DM][DF] -> W1t [e][DF][DM]
  transpose_cvt_kernel<<<dim3(DF / 32, DM / 32, NE), 256, 0, stream>>>(
      W1, W1t, DM, DF);
  // W2 [e][DF][DM] -> W2t [e][DM][DF]
  transpose_cvt_kernel<<<dim3(DM / 32, DF / 32, NE), 256, 0, stream>>>(
      W2, W2t, DF, DM);

  gate_kernel<<<TOKENS / 8, 256, 0, stream>>>(x, Wg, bg, gates);

  const size_t lds_bytes = (size_t)MT * XLD2 * sizeof(f16) +
                           (size_t)2 * MT * HLD2 * sizeof(f16) +
                           (size_t)MT * NE * sizeof(float);  // ~200 KB
  moe_ffn_kernel<<<TOKENS / MT, 512, lds_bytes, stream>>>(x, W1t, b1, W2t, b2,
                                                          gates, out);
  (void)in_sizes; (void)n_in; (void)out_size; (void)ws_size;
}
